// ResNet18_35811437314810
// MI455X (gfx1250) — matrix operand order repaired and checked by reading, not compiled, not run on hardware
//
#include <hip/hip_runtime.h>
#include <cstdint>
#include <cstddef>

// ---------------------------------------------------------------------------
// CDNA5 (gfx1250) implementation of the 3-frame ResNet18 + patch-head model.
// All convolutions are implicit GEMMs on v_wmma_f32_16x16x32_f16 (fp16 in,
// fp32 accumulate), BN / residual / ReLU fused in the epilogue.
// Weights are pre-converted to fp16 (K-padded to x32) once per launch so tile
// staging is pure b128 traffic; B tiles are built as im2col^T with one
// ds_store_b128 per thread per K-step. Each wave computes a 32x32 C tile
// (2 A frags x 2 B frags -> 4 WMMAs per K-step).
// ---------------------------------------------------------------------------

typedef __attribute__((ext_vector_type(16))) _Float16 v16h;
typedef __attribute__((ext_vector_type(8)))  float    v8f;

#define TM 128   // block M tile (4 wave-rows x 32)
#define TN 64    // block N tile (2 wave-cols x 32)
#define TK 32    // K step (WMMA K)
#define EPS 1e-5f

// ---- fragment gathers following the documented CDNA5 WMMA VGPR layouts ----
__device__ __forceinline__ v16h load_a_frag(const _Float16 (*As)[TK], int row0, int lane) {
  // A 16x32 f16: lanes 0-15 kb=0, lanes 16-31 kb=8; VGPR v<4: K=kb+2v; v>=4: K=16+kb+2(v-4)
  union { v16h h; uint32_t u[8]; } f;
  const uint32_t* Ar = reinterpret_cast<const uint32_t*>(&As[row0 + (lane & 15)][0]);
  const int kb2 = (lane < 16) ? 0 : 4;   // dword offset
  #pragma unroll
  for (int v = 0; v < 4; ++v) { f.u[v] = Ar[kb2 + v]; f.u[4 + v] = Ar[8 + kb2 + v]; }
  return f.h;
}
__device__ __forceinline__ v16h load_b_frag(const _Float16 (*Bs)[TK], int col0, int lane) {
  // B 32x16 f16 (stored K-transposed): lanes 0-15 kb=0, lanes 16-31 kb=8; VGPR v<4: K=kb+2v; v>=4: K=16+kb+2(v-4)
  union { v16h h; uint32_t u[8]; } f;
  const uint32_t* Br = reinterpret_cast<const uint32_t*>(&Bs[col0 + (lane & 15)][0]);
  const int ko2 = (lane < 16) ? 0 : 4;
  #pragma unroll
  for (int v = 0; v < 4; ++v) { f.u[v] = Br[ko2 + v]; f.u[4 + v] = Br[8 + ko2 + v]; }
  return f.h;
}

// ---------------------------------------------------------------------------
// Implicit-GEMM conv + BN (+residual) (+ReLU). wgt is fp16, [Cout, Kpad].
// ---------------------------------------------------------------------------
template <int KS>
__global__ __launch_bounds__(256) void conv_bn_wmma(
    const _Float16* __restrict__ in, const _Float16* __restrict__ wgt, int Kpad,
    const float* __restrict__ bn_g, const float* __restrict__ bn_b,
    const float* __restrict__ bn_m, const float* __restrict__ bn_v,
    const _Float16* __restrict__ resid, _Float16* __restrict__ out,
    int Nb, int Cin, int Hin, int Win, int Cout, int Hout, int Wout,
    int stride, int pad, int relu)
{
  constexpr int KK = KS * KS;
  __shared__ alignas(16) _Float16 As[TM][TK];   // weights,  [m][k]   8KB
  __shared__ alignas(16) _Float16 Bs[TN][TK];   // im2col^T, [n][k]   4KB

  const int tid   = threadIdx.x;
  const int wave  = tid >> 5;
  const int lane  = tid & 31;
  const int mfrag = wave >> 1;        // 0..3 -> 32-row slab
  const int nfrag = wave & 1;         // 0..1 -> 32-col slab
  const int m0    = blockIdx.y * TM;
  const int n0    = blockIdx.x * TN;
  const int Kg    = Cin * KK;
  const int HW    = Hout * Wout;
  const int Ng    = Nb * HW;

  v8f acc[2][2];
  #pragma unroll
  for (int a = 0; a < 2; ++a)
    #pragma unroll
    for (int b = 0; b < 2; ++b)
      #pragma unroll
      for (int e = 0; e < 8; ++e) acc[a][b][e] = 0.0f;

  for (int k0 = 0; k0 < Kg; k0 += TK) {
    // ---- A tile: straight fp16 b128 copies (weights pre-padded to Kpad) ----
    #pragma unroll
    for (int idx = tid; idx < TM * 4; idx += 256) {       // 8 halves per chunk
      int mm = idx >> 2, kc = (idx & 3) << 3;
      int m = m0 + mm;
      uint4 q = make_uint4(0u, 0u, 0u, 0u);
      if (m < Cout)
        q = *reinterpret_cast<const uint4*>(wgt + (size_t)m * Kpad + k0 + kc);
      *reinterpret_cast<uint4*>(&As[mm][kc]) = q;
    }
    // ---- B tile: one (n, 8k) chunk per thread, single b128 LDS store ----
    {
      int nn = tid >> 2, kc = (tid & 3) << 3;
      int n = n0 + nn;
      union { uint4 q; _Float16 h[8]; } t;
      t.q = make_uint4(0u, 0u, 0u, 0u);
      if (n < Ng) {
        int b = n / HW, rem = n % HW;
        int oh = rem / Wout, ow = rem % Wout;
        int ihb = oh * stride - pad, iwb = ow * stride - pad;
        const _Float16* inb = in + (size_t)b * Cin * Hin * Win;
        #pragma unroll
        for (int j = 0; j < 8; ++j) {
          int k = k0 + kc + j;
          if (k < Kg) {
            int c  = k / KK;               // KS is compile-time: cheap
            int r2 = k - c * KK;
            int r  = r2 / KS, s = r2 - r * KS;
            int ih = ihb + r, iw = iwb + s;
            if (ih >= 0 && ih < Hin && iw >= 0 && iw < Win)
              t.h[j] = inb[((size_t)c * Hin + ih) * Win + iw];
          }
        }
      }
      *reinterpret_cast<uint4*>(&Bs[nn][kc]) = t.q;
    }
    __syncthreads();

    v16h a0 = load_a_frag(As, mfrag * 32,      lane);
    v16h a1 = load_a_frag(As, mfrag * 32 + 16, lane);
    v16h b0 = load_b_frag(Bs, nfrag * 32,      lane);
    v16h b1 = load_b_frag(Bs, nfrag * 32 + 16, lane);
    acc[0][0] = __builtin_amdgcn_wmma_f32_16x16x32_f16(false, a0, false, b0, (short)0, acc[0][0], false, false);
    acc[0][1] = __builtin_amdgcn_wmma_f32_16x16x32_f16(false, a0, false, b1, (short)0, acc[0][1], false, false);
    acc[1][0] = __builtin_amdgcn_wmma_f32_16x16x32_f16(false, a1, false, b0, (short)0, acc[1][0], false, false);
    acc[1][1] = __builtin_amdgcn_wmma_f32_16x16x32_f16(false, a1, false, b1, (short)0, acc[1][1], false, false);
    __syncthreads();
  }

  // ---- epilogue: C/D layout (VGPR v: M=v lanes 0-15, M=v+8 lanes 16-31) ----
  const int hi8   = (lane < 16) ? 0 : 8;
  const int nlane = lane & 15;
  #pragma unroll
  for (int ai = 0; ai < 2; ++ai) {
    const int mb = m0 + mfrag * 32 + ai * 16 + hi8;
    float inv8[8], bs8[8];
    #pragma unroll
    for (int v = 0; v < 8; ++v) {
      int m = mb + v;
      if (m < Cout) {
        float iv = bn_g[m] * rsqrtf(bn_v[m] + EPS);
        inv8[v] = iv; bs8[v] = bn_b[m] - bn_m[m] * iv;
      } else { inv8[v] = 0.0f; bs8[v] = 0.0f; }
    }
    #pragma unroll
    for (int bi = 0; bi < 2; ++bi) {
      int n = n0 + nfrag * 32 + bi * 16 + nlane;
      if (n >= Ng) continue;
      int b = n / HW, rem = n % HW;
      #pragma unroll
      for (int v = 0; v < 8; ++v) {
        int m = mb + v;
        if (m < Cout) {
          float val = acc[ai][bi][v] * inv8[v] + bs8[v];
          if (resid) val += (float)resid[((size_t)b * Cout + m) * HW + rem];
          if (relu)  val = fmaxf(val, 0.0f);
          out[((size_t)b * Cout + m) * HW + rem] = (_Float16)val;
        }
      }
    }
  }
}

// ---------------------------------------------------------------------------
// Support kernels
// ---------------------------------------------------------------------------
__global__ void w2h_k(const float* __restrict__ w, _Float16* __restrict__ o,
                      int Cout, int Kg, int Kp) {
  int i = blockIdx.x * blockDim.x + threadIdx.x;
  if (i >= Cout * Kp) return;
  int m = i / Kp, k = i % Kp;
  o[i] = (k < Kg) ? (_Float16)w[m * Kg + k] : (_Float16)0.0f;
}

__global__ void extract_frame_k(const float* __restrict__ x,
                                _Float16* __restrict__ out, int total, int frame) {
  int i = blockIdx.x * blockDim.x + threadIdx.x;
  if (i < total) out[i] = (_Float16)x[i * 3 + frame];
}

__global__ void maxpool3s2p1_k(const _Float16* __restrict__ in,
                               _Float16* __restrict__ out,
                               int NC, int Hin, int Win, int Hout, int Wout) {
  int i = blockIdx.x * blockDim.x + threadIdx.x;
  if (i >= NC * Hout * Wout) return;
  int ow = i % Wout, t = i / Wout;
  int oh = t % Hout, nc = t / Hout;
  float m = -3.4e38f;
  #pragma unroll
  for (int r = 0; r < 3; ++r) {
    int ih = oh * 2 - 1 + r;
    if (ih < 0 || ih >= Hin) continue;
    #pragma unroll
    for (int s = 0; s < 3; ++s) {
      int iw = ow * 2 - 1 + s;
      if (iw < 0 || iw >= Win) continue;
      m = fmaxf(m, (float)in[(nc * Hin + ih) * Win + iw]);
    }
  }
  out[i] = (_Float16)m;
}

// patches in merged layout [B*P, C, 7, 7];  points[b, frame, 2p]=x, [..,2p+1]=y
__global__ void gather_patches_k(const _Float16* __restrict__ feat,
                                 const int* __restrict__ pts,
                                 _Float16* __restrict__ out,
                                 int B, int C, int H, int W, int P, int frame) {
  int i = blockIdx.x * blockDim.x + threadIdx.x;
  int total = B * P * C * 49;
  if (i >= total) return;
  int dx = i % 7, t = i / 7;
  int dy = t % 7; t /= 7;
  int c  = t % C; t /= C;
  int p  = t % P;
  int b  = t / P;
  int px = pts[(b * 3 + frame) * (2 * P) + 2 * p]     - 3;  // indexes H
  int py = pts[(b * 3 + frame) * (2 * P) + 2 * p + 1] - 3;  // indexes W
  out[i] = feat[((b * C + c) * H + (px + dy)) * W + (py + dx)];
}

__global__ void sub_k(const _Float16* __restrict__ a, const _Float16* __restrict__ b,
                      _Float16* __restrict__ o, int n) {
  int i = blockIdx.x * blockDim.x + threadIdx.x;
  if (i < n) o[i] = (_Float16)((float)a[i] - (float)b[i]);
}

__global__ void mean_hw_k(const _Float16* __restrict__ in, float* __restrict__ out,
                          int NC, int HW) {
  int i = blockIdx.x * blockDim.x + threadIdx.x;
  if (i >= NC) return;
  float s = 0.0f;
  for (int j = 0; j < HW; ++j) s += (float)in[i * HW + j];
  out[i] = s / (float)HW;
}

__global__ void linear_sigmoid_k(const float* __restrict__ x, const float* __restrict__ w,
                                 const float* __restrict__ b, float* __restrict__ out,
                                 int N, int K) {
  int i = blockIdx.x * blockDim.x + threadIdx.x;
  if (i >= N) return;
  float s = b[0];
  for (int k = 0; k < K; ++k) s += x[i * K + k] * w[k];
  out[i] = 1.0f / (1.0f + expf(-s));
}

__global__ void avg2_k(const float* a, const float* b, float* o, int n) {
  int i = blockIdx.x * blockDim.x + threadIdx.x;
  if (i < n) o[i] = 0.5f * (a[i] + b[i]);
}

// fn[b, c, p] = bx[(b*P+p)*32 + c] * (beta+alpha)/2  ->  featcat[b][c*P + p]
__global__ void assemble_fn_k(const float* __restrict__ bx, const float* __restrict__ beta,
                              const float* __restrict__ alpha, float* __restrict__ featcat,
                              int B, int P, int C32) {
  int i = blockIdx.x * blockDim.x + threadIdx.x;
  if (i >= B * C32 * P) return;
  int b = i / (C32 * P), r = i % (C32 * P);
  int c = r / P, p = r % P;
  int bp = b * P + p;
  featcat[b * 1024 + c * P + p] = bx[bp * C32 + c] * (beta[bp] + alpha[bp]) * 0.5f;
}

__global__ void bf_gate_k(const float* __restrict__ bfm, const float* __restrict__ gam,
                          float* __restrict__ featcat, int B, int C) {
  int i = blockIdx.x * blockDim.x + threadIdx.x;
  if (i >= B * C) return;
  int b = i / C, c = i % C;
  featcat[b * 1024 + 512 + c] = bfm[b * C + c] * gam[b];
}

__global__ void fc7_k(const float* __restrict__ featcat, const float* __restrict__ w,
                      const float* __restrict__ bias, float* __restrict__ out, int B) {
  int i = blockIdx.x * blockDim.x + threadIdx.x;
  if (i >= B * 7) return;
  int b = i / 7, j = i % 7;
  float s = bias[j];
  for (int k = 0; k < 1024; ++k) s += featcat[b * 1024 + k] * w[j * 1024 + k];
  out[b * 7 + j] = s;
}

// ---------------------------------------------------------------------------
// Host-side launch
// ---------------------------------------------------------------------------
struct BNp { const float *g, *b, *m, *v; };
struct Blk { const float* w1; BNp bn1; const float* w2; BNp bn2;
             const float* wd; BNp bnd; bool has_d; };
struct HBlk { const _Float16 *w1, *w2, *wd; BNp bn1, bn2, bnd; bool has_d;
              int kp1, kp2, kpd; };

extern "C" void kernel_launch(void* const* d_in, const int* in_sizes, int n_in,
                              void* d_out, int out_size, void* d_ws, size_t ws_size,
                              hipStream_t stream) {
  (void)in_sizes; (void)n_in; (void)out_size; (void)ws_size;
  const int B = 32, P = 16;

  const float* x_in   = (const float*)d_in[0];
  const int*   points = (const int*)d_in[1];

  // Params flattened depth-first in dict insertion order.
  int cur = 2;
  auto nf  = [&]() { return (const float*)d_in[cur++]; };
  auto nbn = [&]() { BNp r; r.g = nf(); r.b = nf(); r.m = nf(); r.v = nf(); return r; };
  auto nbk = [&](bool down) {
    Blk k; k.w1 = nf(); k.bn1 = nbn(); k.w2 = nf(); k.bn2 = nbn();
    k.has_d = down; k.wd = nullptr;
    if (down) { k.wd = nf(); k.bnd = nbn(); }
    return k;
  };
  const float* conv1 = nf(); BNp bn1 = nbn();
  Blk l1[2] = { nbk(false), nbk(false) };
  Blk l2[2] = { nbk(true),  nbk(false) };
  Blk l3[2] = { nbk(true),  nbk(false) };
  Blk l4[2] = { nbk(true),  nbk(false) };
  const float* pa_w = nf();   BNp pa_bn  = nbn();
  const float* al_w = nf();   const float* al_b = nf();
  const float* pb_w1 = nf();  BNp pb_bn1 = nbn();
  const float* pb_w2 = nf();  BNp pb_bn2 = nbn();
  const float* be_w = nf();   const float* be_b = nf();
  const float* ga_w = nf();   const float* ga_b = nf();
  const float* fc_w = nf();   const float* fc_b = nf();

  // bump allocator on workspace
  char* base = (char*)d_ws;
  size_t off = 0;
  auto alloc = [&](size_t bytes) { void* p = base + off; off += (bytes + 255) & ~(size_t)255; return p; };
  auto h16  = [&](size_t elems) { return (_Float16*)alloc(elems * 2); };
  auto f32  = [&](size_t elems) { return (float*)alloc(elems * 4); };
  auto g1d  = [](int n) { return dim3((n + 255) / 256); };

  // ---- pre-convert all conv weights to fp16, K padded to x32 ----
  auto cvtw = [&](const float* w, int Cout, int Kg) {
    int Kp = (Kg + 31) & ~31;
    _Float16* o = h16((size_t)Cout * Kp);
    w2h_k<<<g1d(Cout * Kp), 256, 0, stream>>>(w, o, Cout, Kg, Kp);
    return o;
  };
  auto cvtblk = [&](const Blk& b, int Cin, int Cout) {
    HBlk h;
    h.w1 = cvtw(b.w1, Cout, Cin * 9);  h.kp1 = (Cin * 9 + 31) & ~31;  h.bn1 = b.bn1;
    h.w2 = cvtw(b.w2, Cout, Cout * 9); h.kp2 = (Cout * 9 + 31) & ~31; h.bn2 = b.bn2;
    h.has_d = b.has_d; h.wd = nullptr; h.kpd = (Cin + 31) & ~31;
    if (b.has_d) { h.wd = cvtw(b.wd, Cout, Cin); h.bnd = b.bnd; }
    return h;
  };
  const _Float16* conv1h = cvtw(conv1, 64, 147);  const int conv1kp = 160;
  HBlk hl1[2] = { cvtblk(l1[0],  64,  64), cvtblk(l1[1],  64,  64) };
  HBlk hl2[2] = { cvtblk(l2[0],  64, 128), cvtblk(l2[1], 128, 128) };
  HBlk hl3[2] = { cvtblk(l3[0], 128, 256), cvtblk(l3[1], 256, 256) };
  HBlk hl4[2] = { cvtblk(l4[0], 256, 512), cvtblk(l4[1], 512, 512) };
  const _Float16* pa_h  = cvtw(pa_w,   32, 1152);
  const _Float16* pb1_h = cvtw(pb_w1, 128, 1152);
  const _Float16* pb2_h = cvtw(pb_w2,  32, 1152);

  // ---- activation buffers (fp16) ----
  _Float16* F   = h16((size_t)B * 3 * 224 * 224);
  _Float16* A1  = h16((size_t)B * 64 * 112 * 112);
  _Float16* P1  = h16((size_t)B * 64 * 56 * 56);
  _Float16* T1  = h16((size_t)B * 64 * 56 * 56);
  _Float16* T2  = h16((size_t)B * 64 * 56 * 56);
  _Float16* D1  = h16((size_t)B * 128 * 28 * 28);
  _Float16* DS  = h16((size_t)B * 128 * 28 * 28);
  _Float16* D2  = h16((size_t)B * 128 * 28 * 28);
  _Float16* feats[3]; for (int f = 0; f < 3; ++f) feats[f] = h16((size_t)B * 128 * 28 * 28);
  _Float16* patch[3]; for (int f = 0; f < 3; ++f) patch[f] = h16((size_t)B * P * 128 * 49);
  _Float16* diff1 = h16((size_t)B * P * 128 * 49);
  _Float16* diff2 = h16((size_t)B * P * 128 * 49);
  _Float16* aco  = h16((size_t)B * P * 32 * 49);
  float* am1  = f32((size_t)B * P * 32);
  float* am2  = f32((size_t)B * P * 32);
  float* a1   = f32((size_t)B * P);
  float* a2   = f32((size_t)B * P);
  float* alph = f32((size_t)B * P);
  _Float16* bco1 = h16((size_t)B * P * 128 * 49);
  _Float16* bco2 = h16((size_t)B * P * 32 * 49);
  float* bx   = f32((size_t)B * P * 32);
  float* betv = f32((size_t)B * P);
  _Float16* E1 = h16((size_t)B * 256 * 14 * 14);
  _Float16* ES = h16((size_t)B * 256 * 14 * 14);
  _Float16* E2 = h16((size_t)B * 256 * 14 * 14);
  _Float16* G1 = h16((size_t)B * 512 * 7 * 7);
  _Float16* GS = h16((size_t)B * 512 * 7 * 7);
  _Float16* G2 = h16((size_t)B * 512 * 7 * 7);
  float* bfm  = f32((size_t)B * 512);
  float* gam  = f32((size_t)B);
  float* fcat = f32((size_t)B * 1024);

  auto conv = [&](const _Float16* in, const _Float16* w, int Kp, BNp bn,
                  const _Float16* res, _Float16* out, int Nb, int Cin, int Hin,
                  int Win, int Cout, int KS, int stride, int pad, int relu) {
    int Ho = (Hin + 2 * pad - KS) / stride + 1;
    int Wo = (Win + 2 * pad - KS) / stride + 1;
    int Ng = Nb * Ho * Wo;
    dim3 grid((Ng + TN - 1) / TN, (Cout + TM - 1) / TM);
    if (KS == 1)
      conv_bn_wmma<1><<<grid, 256, 0, stream>>>(in, w, Kp, bn.g, bn.b, bn.m, bn.v, res, out,
                                                Nb, Cin, Hin, Win, Cout, Ho, Wo, stride, pad, relu);
    else if (KS == 3)
      conv_bn_wmma<3><<<grid, 256, 0, stream>>>(in, w, Kp, bn.g, bn.b, bn.m, bn.v, res, out,
                                                Nb, Cin, Hin, Win, Cout, Ho, Wo, stride, pad, relu);
    else
      conv_bn_wmma<7><<<grid, 256, 0, stream>>>(in, w, Kp, bn.g, bn.b, bn.m, bn.v, res, out,
                                                Nb, Cin, Hin, Win, Cout, Ho, Wo, stride, pad, relu);
  };
  auto run_block = [&](const _Float16* xin, const HBlk& K, int Nb, int Cin, int Hin,
                       int Win, int Cout, int stride, _Float16* t1, _Float16* sc,
                       _Float16* out) {
    int Ho = (Hin + 2 - 3) / stride + 1;
    int Wo = (Win + 2 - 3) / stride + 1;
    conv(xin, K.w1, K.kp1, K.bn1, nullptr, t1, Nb, Cin, Hin, Win, Cout, 3, stride, 1, 1);
    const _Float16* res = xin;
    if (K.has_d) {
      conv(xin, K.wd, K.kpd, K.bnd, nullptr, sc, Nb, Cin, Hin, Win, Cout, 1, stride, 0, 0);
      res = sc;
    }
    conv(t1, K.w2, K.kp2, K.bn2, res, out, Nb, Cout, Ho, Wo, Cout, 3, 1, 1, 1);
  };

  // ---------------- stems (3 frames) ----------------
  for (int f = 0; f < 3; ++f) {
    int tot = B * 3 * 224 * 224;
    extract_frame_k<<<g1d(tot), 256, 0, stream>>>(x_in, F, tot, f);
    conv(F, conv1h, conv1kp, bn1, nullptr, A1, B, 3, 224, 224, 64, 7, 2, 3, 1);
    maxpool3s2p1_k<<<g1d(B * 64 * 56 * 56), 256, 0, stream>>>(A1, P1, B * 64, 112, 112, 56, 56);
    run_block(P1, hl1[0], B, 64, 56, 56, 64, 1, T1, nullptr, T2);
    run_block(T2, hl1[1], B, 64, 56, 56, 64, 1, T1, nullptr, P1);
    run_block(P1, hl2[0], B, 64, 56, 56, 128, 2, D1, DS, D2);
    run_block(D2, hl2[1], B, 128, 28, 28, 128, 1, D1, nullptr, feats[f]);
  }

  // ---------------- patches + diffs ----------------
  int ptot = B * P * 128 * 49;
  for (int f = 0; f < 3; ++f)
    gather_patches_k<<<g1d(ptot), 256, 0, stream>>>(feats[f], points, patch[f],
                                                    B, 128, 28, 28, P, f);
  sub_k<<<g1d(ptot), 256, 0, stream>>>(patch[1], patch[0], diff1, ptot);
  sub_k<<<g1d(ptot), 256, 0, stream>>>(patch[2], patch[1], diff2, ptot);

  // ---------------- alpha head (x2, averaged) ----------------
  conv(diff1, pa_h, 1152, pa_bn, nullptr, aco, B * P, 128, 7, 7, 32, 3, 1, 1, 0);
  mean_hw_k<<<g1d(B * P * 32), 256, 0, stream>>>(aco, am1, B * P * 32, 49);
  linear_sigmoid_k<<<g1d(B * P), 256, 0, stream>>>(am1, al_w, al_b, a1, B * P, 32);
  conv(diff2, pa_h, 1152, pa_bn, nullptr, aco, B * P, 128, 7, 7, 32, 3, 1, 1, 0);
  mean_hw_k<<<g1d(B * P * 32), 256, 0, stream>>>(aco, am2, B * P * 32, 49);
  linear_sigmoid_k<<<g1d(B * P), 256, 0, stream>>>(am2, al_w, al_b, a2, B * P, 32);
  avg2_k<<<g1d(B * P), 256, 0, stream>>>(a1, a2, alph, B * P);

  // ---------------- beta head on patches[2] ----------------
  conv(patch[2], pb1_h, 1152, pb_bn1, nullptr, bco1, B * P, 128, 7, 7, 128, 3, 1, 1, 1);
  conv(bco1, pb2_h, 1152, pb_bn2, nullptr, bco2, B * P, 128, 7, 7, 32, 3, 1, 1, 0);
  mean_hw_k<<<g1d(B * P * 32), 256, 0, stream>>>(bco2, bx, B * P * 32, 49);
  linear_sigmoid_k<<<g1d(B * P), 256, 0, stream>>>(bx, be_w, be_b, betv, B * P, 32);

  // ---------------- layer3 / layer4 on feats[2] ----------------
  run_block(feats[2], hl3[0], B, 128, 28, 28, 256, 2, E1, ES, E2);
  run_block(E2, hl3[1], B, 256, 14, 14, 256, 1, E1, nullptr, ES);   // out -> ES
  run_block(ES, hl4[0], B, 256, 14, 14, 512, 2, G1, GS, G2);
  run_block(G2, hl4[1], B, 512, 7, 7, 512, 1, G1, nullptr, GS);     // out -> GS

  mean_hw_k<<<g1d(B * 512), 256, 0, stream>>>(GS, bfm, B * 512, 49);
  linear_sigmoid_k<<<g1d(B), 256, 0, stream>>>(bfm, ga_w, ga_b, gam, B, 512);

  // ---------------- assemble features + FC ----------------
  assemble_fn_k<<<g1d(B * 32 * P), 256, 0, stream>>>(bx, betv, alph, fcat, B, P, 32);
  bf_gate_k<<<g1d(B * 512), 256, 0, stream>>>(bfm, gam, fcat, B, 512);
  fc7_k<<<g1d(B * 7), 256, 0, stream>>>(fcat, fc_w, fc_b, (float*)d_out, B);
}
